// LightSANsLayer_74921409511699
// MI455X (gfx1250) — matrix operand order repaired and checked by reading, not compiled, not run on hardware
//
#include <hip/hip_runtime.h>

typedef __bf16 bhalf;
typedef __attribute__((ext_vector_type(8)))  bhalf v8bf;
typedef __attribute__((ext_vector_type(16))) bhalf v16bf;
typedef __attribute__((ext_vector_type(8)))  float v8f;

constexpr int Bc = 64, Lc = 512, Dc = 768, Hc = 12, Kc = 16, FFc = 3072, HDc = 64;
constexpr long BLc = (long)Bc * Lc;   // 32768

enum { MODE_STORE = 0, MODE_BIAS = 1, MODE_GELU = 2, MODE_QKV = 3, MODE_ACC = 4, MODE_SCATB = 5 };

__device__ __forceinline__ void wait_asynccnt0()
{
#if __has_builtin(__builtin_amdgcn_s_wait_asynccnt)
    __builtin_amdgcn_s_wait_asynccnt(0);
#else
    asm volatile("s_wait_asynccnt 0x0" ::: "memory");
#endif
}

// ---------------------------------------------------------------------------
// Tiled bf16 WMMA GEMM:  C[M,N] = A[M,K] (row-major, bf16) x Bt[N,K]^T (bf16)
// Block tile 128x128, 8 waves (4x2), each wave 32x64 = 2x4 tiles of 16x16.
// K-step 32, double-buffered LDS filled with GLOBAL_LOAD_ASYNC_TO_LDS_B128.
// M must be a multiple of 128 and K a multiple of 32 (true for all calls).
// N may be ragged (guarded). Optional batching via blockIdx.z + strides.
// ---------------------------------------------------------------------------
__global__ __launch_bounds__(256)
void gemm_bf16_wmma(const bhalf* __restrict__ A, const bhalf* __restrict__ Bt,
                    float* __restrict__ Cf, bhalf* __restrict__ Cb,
                    const float* __restrict__ bias,
                    int M, int N, int K,
                    long sA, long sB, long sC, int aMod, int mode)
{
    const int z = blockIdx.z;
    const bhalf* Ab = A + (long)(aMod > 0 ? (z % aMod) : z) * sA;
    const bhalf* Bb = Bt + (long)z * sB;
    const long cOff = (long)z * sC;

    const int tileM = blockIdx.x * 128;
    const int tileN = blockIdx.y * 128;

    __shared__ bhalf As[2][128][32];   // 2 x 8 KB
    __shared__ bhalf Bs[2][128][32];   // 2 x 8 KB (rows are Bt rows = B columns)

    const int tid  = threadIdx.x;
    const int lane = tid & 31;
    const int wave = tid >> 5;
    const int wm = (wave >> 1) * 32;   // 0,32,64,96
    const int wn = (wave & 1) * 64;    // 0,64
    const int l15 = lane & 15;
    const int lhi = lane >> 4;         // 0 or 1

    // staging slot owned by this thread: 32 bytes of one row
    const int srow = tid >> 1;          // 0..127
    const int scol = (tid & 1) * 16;    // 0/16

    // pre-zero only the Bs rows that are beyond ragged N (async never writes them)
    if (tileN + srow >= N) {
        v8bf zz;
        for (int e = 0; e < 8; e++) zz[e] = (bhalf)0.0f;
        for (int bufz = 0; bufz < 2; bufz++) {
            *(v8bf*)(&Bs[bufz][srow][scol])     = zz;
            *(v8bf*)(&Bs[bufz][srow][scol + 8]) = zz;
        }
    }

    v8f acc[2][4];
#pragma unroll
    for (int i = 0; i < 2; i++)
#pragma unroll
        for (int j = 0; j < 4; j++)
#pragma unroll
            for (int r = 0; r < 8; r++) acc[i][j][r] = 0.0f;

    // async-DMA one 128x32 A tile + (guarded) 128x32 Bt tile into LDS buffer `buf`
    auto stageTile = [&](int buf, int k0) {
        {
            unsigned la = (unsigned)(size_t)(&As[buf][srow][scol]);
            unsigned long long ga =
                (unsigned long long)(size_t)(Ab + (long)(tileM + srow) * K + k0 + scol);
            asm volatile("global_load_async_to_lds_b128 %0, %1, off"
                         :: "v"(la), "v"(ga) : "memory");
            asm volatile("global_load_async_to_lds_b128 %0, %1, off offset:16"
                         :: "v"(la), "v"(ga) : "memory");
        }
        const int n = tileN + srow;
        if (n < N) {
            unsigned lb = (unsigned)(size_t)(&Bs[buf][srow][scol]);
            unsigned long long gb =
                (unsigned long long)(size_t)(Bb + (long)n * K + k0 + scol);
            asm volatile("global_load_async_to_lds_b128 %0, %1, off"
                         :: "v"(lb), "v"(gb) : "memory");
            asm volatile("global_load_async_to_lds_b128 %0, %1, off offset:16"
                         :: "v"(lb), "v"(gb) : "memory");
        }
    };

    stageTile(0, 0);
    wait_asynccnt0();
    __syncthreads();

    const int nk = K >> 5;
    for (int t = 0; t < nk; ++t) {
        const int cur = t & 1;
        if (t + 1 < nk) stageTile(cur ^ 1, (t + 1) << 5);   // overlap DMA with WMMA

        // A fragment (16x32): lanes 0-15 hold K{0..7,16..23}, lanes 16-31 K{8..15,24..31}
        v16bf af[2], bfr[4];
#pragma unroll
        for (int i = 0; i < 2; i++) {
            const int r = wm + 16 * i + l15;
            v8bf lo = *(const v8bf*)(&As[cur][r][lhi * 8]);
            v8bf hi = *(const v8bf*)(&As[cur][r][16 + lhi * 8]);
#pragma unroll
            for (int e = 0; e < 8; e++) { af[i][e] = lo[e]; af[i][8 + e] = hi[e]; }
        }
        // B fragment (32x16): lane holds column (lane&15), K{0..7,16..23} / K{8..15,24..31} like A
#pragma unroll
        for (int j = 0; j < 4; j++) {
            const int n = wn + 16 * j + l15;
            v8bf blo = *(const v8bf*)(&Bs[cur][n][lhi * 8]);
            v8bf bhi = *(const v8bf*)(&Bs[cur][n][16 + lhi * 8]);
#pragma unroll
            for (int e = 0; e < 8; e++) { bfr[j][e] = blo[e]; bfr[j][8 + e] = bhi[e]; }
        }
#pragma unroll
        for (int i = 0; i < 2; i++)
#pragma unroll
            for (int j = 0; j < 4; j++)
                acc[i][j] = __builtin_amdgcn_wmma_f32_16x16x32_bf16(
                    false, af[i], false, bfr[j], (short)0, acc[i][j], false, false);

        wait_asynccnt0();
        __syncthreads();
    }

    // epilogue (C/D layout: VGPR r, lane l -> row = r + 8*(l>>4), col = l&15)
#pragma unroll
    for (int i = 0; i < 2; i++)
#pragma unroll
        for (int j = 0; j < 4; j++)
#pragma unroll
            for (int r = 0; r < 8; r++) {
                const int row = tileM + wm + 16 * i + r + 8 * lhi;
                const int col = tileN + wn + 16 * j + l15;
                if (col >= N) continue;
                const float v = acc[i][j][r];
                if (mode == MODE_STORE) {
                    Cf[cOff + (long)row * N + col] = v;
                } else if (mode == MODE_BIAS) {
                    Cf[cOff + (long)row * N + col] = v + bias[col];
                } else if (mode == MODE_GELU) {
                    const float xg = v + bias[col];
                    const float g = 0.5f * xg * (1.0f + erff(xg * 0.70710678118654752f));
                    Cb[(long)row * N + col] = (bhalf)g;
                } else if (mode == MODE_QKV) {
                    const int b = row >> 9, l = row & 511, h = col >> 6, d = col & 63;
                    const long o = (((long)b * Hc + h) * Lc + l) * HDc + d;
                    Cf[o] = v;
                    if (Cb) Cb[(((long)b * Hc + h) * HDc + d) * Lc + l] = (bhalf)v; // v^T
                } else if (mode == MODE_ACC) {
                    Cf[cOff + (long)row * N + col] += v;
                } else { // MODE_SCATB
                    const int b = row >> 9, l = row & 511, h = col >> 6, d = col & 63;
                    Cb[(((long)b * Hc + h) * Lc + l) * HDc + d] = (bhalf)v;
                }
            }
}

// ---------------------------------------------------------------------------
// Small helper kernels
// ---------------------------------------------------------------------------
__global__ void cvt_bf16_k(const float* __restrict__ in, bhalf* __restrict__ out, long n)
{
    long i = (long)blockIdx.x * 256 + threadIdx.x;
    if (i < n) out[i] = (bhalf)in[i];
}

// Wt[c, r] = W[r, c]  (store weight transposed in bf16 -> feeds B fragments)
__global__ void transpose_to_bf16_k(const float* __restrict__ W, bhalf* __restrict__ Wt,
                                    int rows, int cols, long n)
{
    long i = (long)blockIdx.x * 256 + threadIdx.x;
    if (i >= n) return;
    int r = (int)(i / cols), c = (int)(i % cols);
    Wt[(long)c * rows + r] = (bhalf)W[i];
}

// softmax over L (axis=1) of d_logits[B,L,K] with padding mask
__global__ void dscore_softmax_k(const float* __restrict__ dlog,
                                 const unsigned char* __restrict__ pad,
                                 float* __restrict__ dsc)
{
    const int kk = blockIdx.x, b = blockIdx.y, tid = threadIdx.x;
    __shared__ float red[256];
    float val[2];
#pragma unroll
    for (int j = 0; j < 2; j++) {
        const int l = tid + j * 256;
        float x = dlog[((long)b * Lc + l) * Kc + kk];
        if (pad[(long)b * Lc + l]) x = -1e9f;
        val[j] = x;
    }
    red[tid] = fmaxf(val[0], val[1]); __syncthreads();
    for (int st = 128; st > 0; st >>= 1) { if (tid < st) red[tid] = fmaxf(red[tid], red[tid + st]); __syncthreads(); }
    const float m = red[0]; __syncthreads();
    float e[2]; e[0] = expf(val[0] - m); e[1] = expf(val[1] - m);
    red[tid] = e[0] + e[1]; __syncthreads();
    for (int st = 128; st > 0; st >>= 1) { if (tid < st) red[tid] += red[tid + st]; __syncthreads(); }
    const float inv = 1.0f / red[0];
#pragma unroll
    for (int j = 0; j < 2; j++) {
        const int l = tid + j * 256;
        dsc[((long)b * Lc + l) * Kc + kk] = e[j] * inv;
    }
}

// k_tilde/v_tilde[b,h,k,d] = sum_l dsc[b,l,k] * {k,v}[b,h,l,d]
__global__ void lowrank_k(const float* __restrict__ dsc, const float* __restrict__ kf,
                          const float* __restrict__ vf, float* __restrict__ ktil,
                          float* __restrict__ vtil)
{
    const int kk = blockIdx.x, h = blockIdx.y, b = blockIdx.z, d = threadIdx.x;
    const float* kp = kf + (((long)b * Hc + h) * Lc) * HDc + d;
    const float* vp = vf + (((long)b * Hc + h) * Lc) * HDc + d;
    const float* sp = dsc + (long)b * Lc * Kc + kk;
    float sk = 0.f, sv = 0.f;
    for (int l = 0; l < Lc; l++) {
        const float w = sp[(long)l * Kc];
        sk += w * kp[(long)l * HDc];
        sv += w * vp[(long)l * HDc];
    }
    const long o = (((long)b * Hc + h) * Kc + kk) * HDc + d;
    ktil[o] = sk; vtil[o] = sv;
}

// item-interest attention over K=16 interests; writes context[b,h,l,d]
__global__ void item_attn_k(const float* __restrict__ qf, const float* __restrict__ ktil,
                            const float* __restrict__ vtil, float* __restrict__ ctx)
{
    const int l = blockIdx.x, h = blockIdx.y, b = blockIdx.z, d = threadIdx.x; // 64 threads
    __shared__ float qrow[HDc];
    __shared__ float sc[Kc];
    const long base = (((long)b * Hc + h) * Lc + l) * HDc;
    qrow[d] = qf[base + d];
    __syncthreads();
    if (d < Kc) {
        const float* kt = ktil + (((long)b * Hc + h) * Kc + d) * HDc;
        float s = 0.f;
        for (int e = 0; e < HDc; e++) s += qrow[e] * kt[e];
        sc[d] = s * 0.125f;  // HD^-0.5
    }
    __syncthreads();
    float mx = -1e30f;
#pragma unroll
    for (int k = 0; k < Kc; k++) mx = fmaxf(mx, sc[k]);
    float p[Kc], den = 0.f;
#pragma unroll
    for (int k = 0; k < Kc; k++) { p[k] = expf(sc[k] - mx); den += p[k]; }
    const float* vt = vtil + (((long)b * Hc + h) * Kc) * HDc + d;
    float o = 0.f;
#pragma unroll
    for (int k = 0; k < Kc; k++) o += p[k] * vt[(long)k * HDc];
    ctx[base + d] = o / den;
}

// causal-masked softmax of pos_scores[h,l,:] (applies HD^-0.5), output bf16
__global__ void pos_softmax_k(const float* __restrict__ ps, const unsigned char* __restrict__ causal,
                              bhalf* __restrict__ pp)
{
    const int l = blockIdx.x, h = blockIdx.y, tid = threadIdx.x;
    __shared__ float red[256];
    const long base = ((long)h * Lc + l) * Lc;
    float val[2];
#pragma unroll
    for (int j = 0; j < 2; j++) {
        const int m = tid + j * 256;
        float x = ps[base + m] * 0.125f;
        if (!causal[(long)l * Lc + m]) x = -1e9f;
        val[j] = x;
    }
    red[tid] = fmaxf(val[0], val[1]); __syncthreads();
    for (int st = 128; st > 0; st >>= 1) { if (tid < st) red[tid] = fmaxf(red[tid], red[tid + st]); __syncthreads(); }
    const float m = red[0]; __syncthreads();
    float e[2]; e[0] = expf(val[0] - m); e[1] = expf(val[1] - m);
    red[tid] = e[0] + e[1]; __syncthreads();
    for (int st = 128; st > 0; st >>= 1) { if (tid < st) red[tid] += red[tid + st]; __syncthreads(); }
    const float inv = 1.0f / red[0];
#pragma unroll
    for (int j = 0; j < 2; j++) {
        const int mm = tid + j * 256;
        pp[base + mm] = (bhalf)(e[j] * inv);
    }
}

// [B,H,L,HD] f32 -> [B,L,D] bf16
__global__ void permute_ctx_k(const float* __restrict__ ctx, bhalf* __restrict__ out, long n)
{
    long i = (long)blockIdx.x * 256 + threadIdx.x;
    if (i >= n) return;
    const int c = (int)(i % Dc);
    const long r = i / Dc;
    const int b = (int)(r >> 9), l = (int)(r & 511);
    const int h = c >> 6, d = c & 63;
    out[i] = (bhalf)ctx[(((long)b * Hc + h) * Lc + l) * HDc + d];
}

// h = LN(a + res) * sc + bi; optionally also write bf16 copy
__global__ void layernorm_k(const float* __restrict__ a, const float* __restrict__ res,
                            const float* __restrict__ sc, const float* __restrict__ bi,
                            float* __restrict__ oF, bhalf* __restrict__ oB)
{
    const long row = blockIdx.x;
    const int tid = threadIdx.x;
    __shared__ float red[256];
    const long base = row * Dc;
    float v[3];
#pragma unroll
    for (int j = 0; j < 3; j++) { const int c = tid + j * 256; v[j] = a[base + c] + res[base + c]; }
    red[tid] = v[0] + v[1] + v[2]; __syncthreads();
    for (int st = 128; st > 0; st >>= 1) { if (tid < st) red[tid] += red[tid + st]; __syncthreads(); }
    const float mu = red[0] * (1.0f / Dc); __syncthreads();
    float q = 0.f;
#pragma unroll
    for (int j = 0; j < 3; j++) { const float dd = v[j] - mu; q += dd * dd; }
    red[tid] = q; __syncthreads();
    for (int st = 128; st > 0; st >>= 1) { if (tid < st) red[tid] += red[tid + st]; __syncthreads(); }
    const float rs = rsqrtf(red[0] * (1.0f / Dc) + 1e-12f);
#pragma unroll
    for (int j = 0; j < 3; j++) {
        const int c = tid + j * 256;
        const float y = (v[j] - mu) * rs * sc[c] + bi[c];
        oF[base + c] = y;
        if (oB) oB[base + c] = (bhalf)y;
    }
}

// ---------------------------------------------------------------------------
extern "C" void kernel_launch(void* const* d_in, const int* in_sizes, int n_in,
                              void* d_out, int out_size, void* d_ws, size_t ws_size,
                              hipStream_t stream)
{
    (void)in_sizes; (void)n_in; (void)out_size; (void)ws_size;
    const float* x        = (const float*)d_in[0];
    const float* pos_emb  = (const float*)d_in[1];
    const unsigned char* pad    = (const unsigned char*)d_in[2];
    const unsigned char* causal = (const unsigned char*)d_in[3];
    const float* w_theta  = (const float*)d_in[4];
    const float* w_q      = (const float*)d_in[5];
    const float* w_k      = (const float*)d_in[6];
    const float* w_v      = (const float*)d_in[7];
    const float* w_pos_q  = (const float*)d_in[8];
    const float* w_pos_k  = (const float*)d_in[9];
    const float* w_out    = (const float*)d_in[10];
    const float* b_out    = (const float*)d_in[11];
    const float* w_ffn1   = (const float*)d_in[12];
    const float* b_ffn1   = (const float*)d_in[13];
    const float* w_ffn2   = (const float*)d_in[14];
    const float* b_ffn2   = (const float*)d_in[15];
    const float* ln1_s    = (const float*)d_in[16];
    const float* ln1_b    = (const float*)d_in[17];
    const float* ln2_s    = (const float*)d_in[18];
    const float* ln2_b    = (const float*)d_in[19];
    float* out = (float*)d_out;

    char* wsb = (char*)d_ws;
    size_t off = 0;
    auto take = [&](size_t bytes) -> char* {
        char* p = wsb + off;
        off += (bytes + 255) & ~(size_t)255;
        return p;
    };
    bhalf* xb    = (bhalf*)take(BLc * Dc * 2);
    bhalf* peb   = (bhalf*)take((size_t)Lc * Dc * 2);
    bhalf* wq_t  = (bhalf*)take((size_t)Dc * Dc * 2);
    bhalf* wk_t  = (bhalf*)take((size_t)Dc * Dc * 2);
    bhalf* wv_t  = (bhalf*)take((size_t)Dc * Dc * 2);
    bhalf* wpq_t = (bhalf*)take((size_t)Dc * Dc * 2);
    bhalf* wpk_t = (bhalf*)take((size_t)Dc * Dc * 2);
    bhalf* wo_t  = (bhalf*)take((size_t)Dc * Dc * 2);
    bhalf* wf1_t = (bhalf*)take((size_t)Dc * FFc * 2);
    bhalf* wf2_t = (bhalf*)take((size_t)FFc * Dc * 2);
    bhalf* wth_t = (bhalf*)take((size_t)Kc * Dc * 2);
    float* qf    = (float*)take(BLc * Dc * 4);
    float* kf    = (float*)take(BLc * Dc * 4);   // contiguous after qf (sizes are 256B multiples)
    float* vf    = (float*)take(BLc * Dc * 4);
    bhalf* vtb   = (bhalf*)take(BLc * Dc * 2);   // v transposed [B,H,HD,L]
    float* dlog  = (float*)take(BLc * Kc * 4);
    float* dsc   = (float*)take(BLc * Kc * 4);
    float* ktil  = (float*)take((size_t)Bc * Hc * Kc * HDc * 4);
    float* vtil  = (float*)take((size_t)Bc * Hc * Kc * HDc * 4);
    float* ctxf  = (float*)take(BLc * Dc * 4);   // context [B,H,L,HD]
    bhalf* pqb   = (bhalf*)take((size_t)Hc * Lc * HDc * 2);
    bhalf* pkb   = (bhalf*)take((size_t)Hc * Lc * HDc * 2);
    float* pscr  = (float*)take((size_t)Hc * Lc * Lc * 4);
    bhalf* pprob = (bhalf*)take((size_t)Hc * Lc * Lc * 2);
    bhalf* ctx_b = (bhalf*)take(BLc * Dc * 2);
    float* hbuf  = (float*)take(BLc * Dc * 4);
    bhalf* hb    = (bhalf*)take(BLc * Dc * 2);
    // aliases (lifetimes verified against the launch order below):
    bhalf* act      = (bhalf*)qf;   // [BL,FF] bf16 == qf+kf footprint; q/k dead by then
    float* ffn_out  = vf;           // v (f32) dead after lowrank einsum
    float* attn_out = ctxf;         // ctx dead once ctx_b is written

    auto cvt = [&](const float* in, bhalf* o, long n) {
        cvt_bf16_k<<<dim3((unsigned)((n + 255) / 256)), 256, 0, stream>>>(in, o, n);
    };
    auto tr = [&](const float* W, bhalf* Wt, int rows, int cols) {
        long n = (long)rows * cols;
        transpose_to_bf16_k<<<dim3((unsigned)((n + 255) / 256)), 256, 0, stream>>>(W, Wt, rows, cols, n);
    };
    auto gemm = [&](const bhalf* A, const bhalf* Bt, float* Cf, bhalf* Cb, const float* bias,
                    int M, int N, int K, long sA, long sB, long sC, int batch, int aMod, int mode) {
        dim3 g((unsigned)(M / 128), (unsigned)((N + 127) / 128), (unsigned)batch);
        gemm_bf16_wmma<<<g, 256, 0, stream>>>(A, Bt, Cf, Cb, bias, M, N, K, sA, sB, sC, aMod, mode);
    };

    // 1. precision conversions
    cvt(x, xb, BLc * Dc);
    cvt(pos_emb, peb, (long)Lc * Dc);
    tr(w_q, wq_t, Dc, Dc);      tr(w_k, wk_t, Dc, Dc);      tr(w_v, wv_t, Dc, Dc);
    tr(w_pos_q, wpq_t, Dc, Dc); tr(w_pos_k, wpk_t, Dc, Dc); tr(w_out, wo_t, Dc, Dc);
    tr(w_ffn1, wf1_t, Dc, FFc); tr(w_ffn2, wf2_t, FFc, Dc); tr(w_theta, wth_t, Dc, Kc);

    // 2. q/k/v projections (scatter to [B,H,L,HD]; v also transposed bf16)
    gemm(xb, wq_t, qf, nullptr, nullptr, (int)BLc, Dc, Dc, 0, 0, 0, 1, 0, MODE_QKV);
    gemm(xb, wk_t, kf, nullptr, nullptr, (int)BLc, Dc, Dc, 0, 0, 0, 1, 0, MODE_QKV);
    gemm(xb, wv_t, vf, vtb,    nullptr, (int)BLc, Dc, Dc, 0, 0, 0, 1, 0, MODE_QKV);

    // 3. interest logits + softmax over L
    gemm(xb, wth_t, dlog, nullptr, nullptr, (int)BLc, Kc, Dc, 0, 0, 0, 1, 0, MODE_STORE);
    dscore_softmax_k<<<dim3(Kc, Bc), 256, 0, stream>>>(dlog, pad, dsc);

    // 4. low-rank keys/values + item-interest attention -> ctxf
    lowrank_k<<<dim3(Kc, Hc, Bc), 64, 0, stream>>>(dsc, kf, vf, ktil, vtil);
    item_attn_k<<<dim3(Lc, Hc, Bc), 64, 0, stream>>>(qf, ktil, vtil, ctxf);

    // 5. decoupled positional attention
    gemm(peb, wpq_t, nullptr, pqb, nullptr, Lc, Dc, Dc, 0, 0, 0, 1, 0, MODE_SCATB);
    gemm(peb, wpk_t, nullptr, pkb, nullptr, Lc, Dc, Dc, 0, 0, 0, 1, 0, MODE_SCATB);
    gemm(pqb, pkb, pscr, nullptr, nullptr, Lc, Lc, HDc,
         (long)Lc * HDc, (long)Lc * HDc, (long)Lc * Lc, Hc, 0, MODE_STORE);
    pos_softmax_k<<<dim3(Lc, Hc), 256, 0, stream>>>(pscr, causal, pprob);
    // pos_context += pos_probs[h] @ v[b,h] (batched over b*H+h; A shared per head)
    gemm(pprob, vtb, ctxf, nullptr, nullptr, Lc, HDc, Lc,
         (long)Lc * Lc, (long)HDc * Lc, (long)Lc * HDc, Bc * Hc, Hc, MODE_ACC);

    // 6. output projection + LN1
    permute_ctx_k<<<dim3((unsigned)((BLc * Dc + 255) / 256)), 256, 0, stream>>>(ctxf, ctx_b, BLc * Dc);
    gemm(ctx_b, wo_t, attn_out, nullptr, b_out, (int)BLc, Dc, Dc, 0, 0, 0, 1, 0, MODE_BIAS);
    layernorm_k<<<dim3((unsigned)BLc), 256, 0, stream>>>(attn_out, x, ln1_s, ln1_b, hbuf, hb);

    // 7. FFN (GELU fused into GEMM1 epilogue, bf16 activations) + LN2
    gemm(hb, wf1_t, nullptr, act, b_ffn1, (int)BLc, FFc, Dc, 0, 0, 0, 1, 0, MODE_GELU);
    gemm(act, wf2_t, ffn_out, nullptr, b_ffn2, (int)BLc, Dc, FFc, 0, 0, 0, 1, 0, MODE_BIAS);
    layernorm_k<<<dim3((unsigned)BLc), 256, 0, stream>>>(ffn_out, hbuf, ln2_s, ln2_b, out, nullptr);
}
